// SparseGatingNetwork_32384053412170
// MI455X (gfx1250) — matrix operand order repaired and checked by reading, not compiled, not run on hardware
//
#include <hip/hip_runtime.h>
#include <math.h>

typedef _Float16 v16h __attribute__((ext_vector_type(16)));
typedef float    v8f  __attribute__((ext_vector_type(8)));
typedef float    v4f  __attribute__((ext_vector_type(4)));

#define NROWS          1048576
#define DDIM           128
#define NEXP           64
#define ROWS_PER_BLOCK 128
#define THREADS        256

__device__ __forceinline__ bool better(float av, int ai, float bv, int bi) {
    // strict ordering: larger value wins; on exact tie, lower expert index wins
    return (av > bv) || (av == bv && ai < bi);
}

__global__ __launch_bounds__(THREADS)
void moe_gate_kernel(const float* __restrict__ x,
                     const float* __restrict__ Wexp,
                     const float* __restrict__ bias,
                     const float* __restrict__ alphap,
                     float* __restrict__ outw,
                     float* __restrict__ outi)
{
    // B operands pre-swizzled: [e(4)][kc(4)][lane(32)] -> 16 halfs (32B) per slot
    __shared__ v16h sBhi[512];
    __shared__ v16h sBlo[512];

    const int tid  = threadIdx.x;
    const int lane = tid & 31;
    const int wave = tid >> 5;
    const int n    = lane & 15;        // column within 16-wide tile
    const int hi16 = (lane >> 4) & 1;  // upper half of wave

    // ---- Stage W into LDS as split-precision f16 in B-operand layout ----
    {
        _Float16* bh = (_Float16*)sBhi;
        _Float16* bl = (_Float16*)sBlo;
        for (int it = 0; it < 32; ++it) {
            int i  = it * THREADS + tid;     // 0..8191 half-slots
            int h  = i & 15;                 // half index within lane slot
            int s  = (i >> 4) & 31;          // lane slot
            int kc = (i >> 9) & 3;           // K chunk
            int e  = (i >> 11) & 3;          // expert tile
            int nn = e * 16 + (s & 15);      // expert (B column)
            int k  = kc * 32 + ((s >> 4) << 3) + (h & 7) + ((h >> 3) << 4);  // lanes 0-15: K 0..7,16..23; 16-31: K 8..15,24..31
            float v = Wexp[nn * DDIM + k];
            _Float16 vh = (_Float16)v;
            _Float16 vl = (_Float16)(v - (float)vh);
            bh[i] = vh;
            bl[i] = vl;
        }
    }
    __syncthreads();

    const float alpha = alphap[0];
    float ab[4];
#pragma unroll
    for (int e = 0; e < 4; ++e) ab[e] = alpha * bias[e * 16 + n];

    const int r0   = blockIdx.x * ROWS_PER_BLOCK + wave * 16;
    const int rowA = r0 + n;                 // A-operand row for this lane
    const int sh   = hi16 ? 8 : 0;           // K shift for upper-half lanes
    const float* xr = x + (size_t)rowA * DDIM;

    v8f acc[4];
#pragma unroll
    for (int e = 0; e < 4; ++e) acc[e] = (v8f){};

#pragma unroll
    for (int kc = 0; kc < 4; ++kc) {
        const int kb = kc * 32;
        // A layout (16-bit, 16x32): lane<16 -> K kb+{0..7, 16..23}; lane>=16 -> +8
        v4f p0 = *(const v4f*)(xr + kb + sh);
        v4f p1 = *(const v4f*)(xr + kb + sh + 4);
        v4f p2 = *(const v4f*)(xr + kb + 16 + sh);
        v4f p3 = *(const v4f*)(xr + kb + 16 + sh + 4);
        float fs[16];
#pragma unroll
        for (int i = 0; i < 4; ++i) { fs[i] = p0[i]; fs[4 + i] = p1[i]; fs[8 + i] = p2[i]; fs[12 + i] = p3[i]; }

        v16h ahi, alo;
#pragma unroll
        for (int i = 0; i < 16; ++i) {
            float v = fs[i];
            _Float16 vh = (_Float16)v;     // RTNE high part
            ahi[i] = vh;
            alo[i] = (_Float16)(v - (float)vh);   // residual
        }

#pragma unroll
        for (int e = 0; e < 4; ++e) {
            v16h bhi = sBhi[(e * 4 + kc) * 32 + lane];
            v16h blo = sBlo[(e * 4 + kc) * 32 + lane];
            acc[e] = __builtin_amdgcn_wmma_f32_16x16x32_f16(false, ahi, false, bhi, (short)0, acc[e], false, false);
            acc[e] = __builtin_amdgcn_wmma_f32_16x16x32_f16(false, alo, false, bhi, (short)0, acc[e], false, false);
            acc[e] = __builtin_amdgcn_wmma_f32_16x16x32_f16(false, ahi, false, blo, (short)0, acc[e], false, false);
        }
    }

    // ---- Epilogue: per-row top-2 over 64 experts, sigmoid, renorm, scatter ----
#pragma unroll
    for (int j = 0; j < 8; ++j) {
        const int row = r0 + j + hi16 * 8;   // C layout: VGPR j -> M=j (lo half) / j+8 (hi half)
        float s0 = acc[0][j] + ab[0];
        float s1 = acc[1][j] + ab[1];
        float s2 = acc[2][j] + ab[2];
        float s3 = acc[3][j] + ab[3];

        float v0 = s0; int i0 = n;
        float v1 = -3.402823466e38f; int i1 = 0x7fffffff;
        if (s1 > v0) { v1 = v0; i1 = i0; v0 = s1; i0 = 16 + n; } else if (s1 > v1) { v1 = s1; i1 = 16 + n; }
        if (s2 > v0) { v1 = v0; i1 = i0; v0 = s2; i0 = 32 + n; } else if (s2 > v1) { v1 = s2; i1 = 32 + n; }
        if (s3 > v0) { v1 = v0; i1 = i0; v0 = s3; i0 = 48 + n; } else if (s3 > v1) { v1 = s3; i1 = 48 + n; }

        // merge sorted pairs across the 16-lane row group
#pragma unroll
        for (int m = 1; m < 16; m <<= 1) {
            float ov0 = __shfl_xor(v0, m, 16);
            float ov1 = __shfl_xor(v1, m, 16);
            int   oi0 = __shfl_xor(i0, m, 16);
            int   oi1 = __shfl_xor(i1, m, 16);
            if (better(ov0, oi0, v0, i0)) {
                if (better(v0, i0, ov1, oi1)) { v1 = v0;  i1 = i0;  }
                else                          { v1 = ov1; i1 = oi1; }
                v0 = ov0; i0 = oi0;
            } else if (better(ov0, oi0, v1, i1)) {
                v1 = ov0; i1 = oi0;
            }
        }

        float w0 = 1.0f / (1.0f + __expf(-v0));
        float w1 = 1.0f / (1.0f + __expf(-v1));
        float inv = 1.0f / (w0 + w1 + 1e-6f);
        w0 *= inv; w1 *= inv;

        const size_t base = (size_t)row * NEXP;
#pragma unroll
        for (int e = 0; e < 4; ++e) {
            const int expert = e * 16 + n;
            float val = (expert == i0) ? w0 : ((expert == i1) ? w1 : 0.0f);
            outw[base + expert] = val;
        }
        if (n == 0) {
            outi[(size_t)row * 2 + 0] = (float)i0;
            outi[(size_t)row * 2 + 1] = (float)i1;
        }
    }
}

extern "C" void kernel_launch(void* const* d_in, const int* in_sizes, int n_in,
                              void* d_out, int out_size, void* d_ws, size_t ws_size,
                              hipStream_t stream) {
    // setup_inputs order: x, edge_index(int64, unused), ppr_values(unused),
    //                     expert_vector, bias, alpha
    const float* x     = (const float*)d_in[0];
    const float* Wexp  = (const float*)d_in[3];
    const float* bias  = (const float*)d_in[4];
    const float* alpha = (const float*)d_in[5];

    float* outw = (float*)d_out;                       // sparse_weights [N,64]
    float* outi = outw + (size_t)NROWS * NEXP;         // top_k_indices [N,2] (as float)

    dim3 grid(NROWS / ROWS_PER_BLOCK);   // 8192 workgroups, 128 rows each
    moe_gate_kernel<<<grid, THREADS, 0, stream>>>(x, Wexp, bias, alpha, outw, outi);
}
